// AdaptiveOutput_26362509263280
// MI455X (gfx1250) — matrix operand order repaired and checked by reading, not compiled, not run on hardware
//
#include <hip/hip_runtime.h>

// ---------------------------------------------------------------------------
// Adaptive log-softmax (adaptive softmax w/ loss) for MI455X (gfx1250).
// bf16 WMMA (v_wmma_f32_16x16x32_bf16) + fused online row-softmax reduction.
// Round 2: compile-time K, hoisted A fragments for tail GEMMs, 1-exp online
// softmax update, head-only captures templated out of the tail hot loops.
// ---------------------------------------------------------------------------

typedef __attribute__((ext_vector_type(16))) __bf16 v16bf;
typedef __attribute__((ext_vector_type(8)))  float  v8f;

constexpr int kN    = 4096;
constexpr int kD    = 1024;
constexpr int kV    = 50257;
constexpr int kCut0 = 2000;
constexpr int kCut1 = 10000;

constexpr int kHeadCols    = kCut0 + 2;      // 2002
constexpr int kHeadColsPad = 2016;           // pad to x16
constexpr int kT0Cols      = kCut1 - kCut0;  // 8000 (already x16)
constexpr int kT0ColsPad   = 8000;
constexpr int kT1Cols      = kV - kCut1;     // 40257
constexpr int kT1ColsPad   = 40272;          // pad to x16
constexpr int kK0 = 256;   // proj0 rank
constexpr int kK1 = 64;    // proj1 rank
constexpr int kMaxChunk = 32;
constexpr int kWaves = 4;  // waves per block in GEMM kernels (blockDim = 128)

__device__ __forceinline__ int imin(int a, int b) { return a < b ? a : b; }
__device__ __forceinline__ int imax(int a, int b) { return a > b ? a : b; }

// ----- WMMA fragment loaders (bf16, 16x16x32) ------------------------------
// A (16x32, MxK), row-major source with leading dim ld:
//   lane L: M = L%16, half = L/16; element e: K = (e%8) + 16*(e/8) + 8*half
//   -> two contiguous 16B chunks at kBase + 8*half and kBase + 16 + 8*half.
__device__ __forceinline__ v16bf load_frag_A(const __bf16* __restrict__ base,
                                             int rowBase, int kBase, int ld,
                                             int lane) {
  int m = lane & 15, half = lane >> 4;
  const __bf16* p = base + (size_t)(rowBase + m) * ld + kBase + half * 8;
  union { uint4 u[2]; v16bf v; } x;
  x.u[0] = *(const uint4*)(p);
  x.u[1] = *(const uint4*)(p + 16);
  return x.v;
}

// B (32x16, KxN) from row-major weight W[col][k] (leading dim ld):
//   lane L: N = L%16, half = L/16; element e: K = (e%8) + 16*(e/8) + 8*half
//   -> two contiguous 16B chunks at kBase + 8*half and kBase + 16 + 8*half.
__device__ __forceinline__ v16bf load_frag_B(const __bf16* __restrict__ W,
                                             int colBase, int kBase, int ld,
                                             int lane) {
  int n = lane & 15, half = lane >> 4;
  const __bf16* p = W + (size_t)(colBase + n) * ld + kBase + half * 8;
  union { uint4 u[2]; v16bf v; } x;
  x.u[0] = *(const uint4*)(p);
  x.u[1] = *(const uint4*)(p + 16);
  return x.v;
}

// ----- f32 -> bf16 convert with zero padding -------------------------------
__global__ void cvt_pad_bf16_kernel(const float* __restrict__ src,
                                    __bf16* __restrict__ dst,
                                    long src_n, long dst_n) {
  long i = (long)blockIdx.x * blockDim.x + threadIdx.x;
  if (i < dst_n) dst[i] = (i < src_n) ? (__bf16)src[i] : (__bf16)0.0f;
}

// ----- per-row capture-column precompute -----------------------------------
__global__ void prep_targets_kernel(const int* __restrict__ tgt,
                                    int* __restrict__ capH,
                                    int* __restrict__ cap0,
                                    int* __restrict__ cap1) {
  int r = blockIdx.x * blockDim.x + threadIdx.x;
  if (r >= kN) return;
  int t  = tgt[r];
  int tc = imin(imax(t, 0), kV - 1);
  capH[r] = imin(tc, kCut0 - 1);
  cap0[r] = imin(imax(tc - kCut0, 0), kCut1 - kCut0 - 1);
  cap1[r] = imin(imax(tc - kCut1, 0), kV - kCut1 - 1);
}

// ----- plain WMMA GEMM: C(bf16) = A(bf16) x W(bf16)^T ----------------------
// grid.x = row tiles (kN/16), grid.y * kWaves + wave = column tile
template <int K>
__global__ void proj_gemm_kernel(const __bf16* __restrict__ A,
                                 const __bf16* __restrict__ W,
                                 __bf16* __restrict__ C, int outCols) {
  int lane = threadIdx.x & 31;
  int wave = threadIdx.x >> 5;
  int rowBase = blockIdx.x * 16;
  int colBase = (blockIdx.y * kWaves + wave) * 16;
  if (colBase >= outCols) return;

  v8f acc = {};
  for (int k = 0; k < K; k += 32) {
    v16bf a = load_frag_A(A, rowBase, k, K, lane);
    v16bf b = load_frag_B(W, colBase, k, K, lane);
    acc = __builtin_amdgcn_wmma_f32_16x16x32_bf16(
        false, a, false, b, (short)0, acc, false, false);
  }
  int n = lane & 15, half = lane >> 4;
#pragma unroll
  for (int v = 0; v < 8; ++v) {
    int row = rowBase + v + 8 * half;
    C[(size_t)row * outCols + colBase + n] = (__bf16)acc[v];
  }
}

// ----- fused WMMA GEMM + online row softmax partials -----------------------
// grid.x = row tile (kN/16), grid.y = column chunk; block = kWaves*32 threads.
// Each wave sweeps col tiles of the chunk round-robin, keeping per-lane
// running (max, sumexp) accumulators per C-slot; merged via LDS at the end.
// HOIST: keep the entire A row-panel (K/32 fragments) in VGPRs across the
// column sweep (used for K<=256); halves VMEM issue per WMMA.
template <int K, bool HOIST, bool HEAD>
__global__ void gemm_softmax_kernel(const __bf16* __restrict__ A,
                                    const __bf16* __restrict__ W,
                                    int ncols, int ntiles, int tilesPerChunk,
                                    const int* __restrict__ capcol,
                                    float* __restrict__ capout,
                                    float* __restrict__ zc0,   // head only
                                    float* __restrict__ zc1,   // head only
                                    float* __restrict__ part_m,
                                    float* __restrict__ part_s) {
  __shared__ float lm[kWaves * 256];
  __shared__ float ls[kWaves * 256];

  constexpr int kFrags = K / 32;

  int lane = threadIdx.x & 31;
  int wave = threadIdx.x >> 5;
  int rowBase = blockIdx.x * 16;
  int chunk = blockIdx.y;
  int tBeg = chunk * tilesPerChunk;
  int tEnd = imin(tBeg + tilesPerChunk, ntiles);

  int n = lane & 15, half = lane >> 4;

  float rm[8], rs[8];
  int capc[8];
#pragma unroll
  for (int v = 0; v < 8; ++v) {
    rm[v] = -INFINITY;
    rs[v] = 0.0f;
    capc[v] = capcol[rowBase + v + 8 * half];
  }

  v16bf afrag[HOIST ? kFrags : 1];
  if (HOIST) {
#pragma unroll
    for (int kk = 0; kk < kFrags; ++kk)
      afrag[kk] = load_frag_A(A, rowBase, kk * 32, K, lane);
  }

  for (int tile = tBeg + wave; tile < tEnd; tile += kWaves) {
    int colBase = tile * 16;
    v8f acc = {};
    if (HOIST) {
#pragma unroll
      for (int kk = 0; kk < kFrags; ++kk) {
        v16bf b = load_frag_B(W, colBase, kk * 32, K, lane);
        acc = __builtin_amdgcn_wmma_f32_16x16x32_bf16(
            false, afrag[kk], false, b, (short)0, acc, false, false);
      }
    } else {
      for (int k = 0; k < K; k += 32) {
        v16bf a = load_frag_A(A, rowBase, k, K, lane);
        v16bf b = load_frag_B(W, colBase, k, K, lane);
        acc = __builtin_amdgcn_wmma_f32_16x16x32_bf16(
            false, a, false, b, (short)0, acc, false, false);
      }
    }
    int gcol = colBase + n;
    bool valid = gcol < ncols;
#pragma unroll
    for (int v = 0; v < 8; ++v) {
      float x = acc[v];
      if (valid) {
        // one-exp online softmax update (rescale only on new max)
        if (x <= rm[v]) {
          rs[v] += __expf(x - rm[v]);
        } else {
          rs[v] = rs[v] * __expf(rm[v] - x) + 1.0f;
          rm[v] = x;
        }
        int grow = rowBase + v + 8 * half;
        if (gcol == capc[v]) capout[grow] = x;
        if (HEAD) {
          if (gcol == kCut0) zc0[grow] = x;
          if (gcol == kCut0 + 1) zc1[grow] = x;
        }
      }
    }
  }

  // stash per-lane partials: [wave][row 0..15][lanecol 0..15]
#pragma unroll
  for (int v = 0; v < 8; ++v) {
    int row = v + 8 * half;
    lm[wave * 256 + row * 16 + n] = rm[v];
    ls[wave * 256 + row * 16 + n] = rs[v];
  }
  __syncthreads();

  if (threadIdx.x < 16) {
    int row = threadIdx.x;
    float m = -INFINITY, s = 0.0f;
    for (int w = 0; w < kWaves; ++w) {
      for (int c = 0; c < 16; ++c) {
        float m2 = lm[w * 256 + row * 16 + c];
        float s2 = ls[w * 256 + row * 16 + c];
        if (m2 == -INFINITY) continue;
        if (m2 <= m) {
          s += s2 * __expf(m2 - m);
        } else {
          s = s * __expf(m - m2) + s2;
          m = m2;
        }
      }
    }
    size_t idx = (size_t)(rowBase + row) * kMaxChunk + chunk;
    part_m[idx] = m;
    part_s[idx] = s;
  }
}

// ----- finalize: merge chunk partials, assemble log-probs, partial sums ----
__device__ __forceinline__ float merge_logz(const float* __restrict__ pm,
                                            const float* __restrict__ ps,
                                            int row, int nchunk) {
  float m = -INFINITY, s = 0.0f;
  for (int c = 0; c < nchunk; ++c) {
    float m2 = pm[(size_t)row * kMaxChunk + c];
    float s2 = ps[(size_t)row * kMaxChunk + c];
    if (m2 == -INFINITY) continue;
    if (m2 <= m) {
      s += s2 * __expf(m2 - m);
    } else {
      s = s * __expf(m - m2) + s2;
      m = m2;
    }
  }
  return m + __logf(s);
}

__global__ void finalize_kernel(const int* __restrict__ tgt,
                                const float* __restrict__ capH,
                                const float* __restrict__ z0,
                                const float* __restrict__ z1,
                                const float* __restrict__ cap0v,
                                const float* __restrict__ cap1v,
                                const float* __restrict__ pmH,
                                const float* __restrict__ psH, int ncH,
                                const float* __restrict__ pm0,
                                const float* __restrict__ ps0, int nc0,
                                const float* __restrict__ pm1,
                                const float* __restrict__ ps1, int nc1,
                                float* __restrict__ out,
                                float* __restrict__ blkSum,
                                int* __restrict__ blkCnt) {
  __shared__ float ssum[256];
  __shared__ int scnt[256];
  int tid = threadIdx.x;
  int r = blockIdx.x * 256 + tid;

  float lzH = merge_logz(pmH, psH, r, ncH);
  float lz0 = merge_logz(pm0, ps0, r, nc0);
  float lz1 = merge_logz(pm1, ps1, r, nc1);

  int t = tgt[r];
  float val = 0.0f;
  int valid = 0;
  if (t >= 0) {
    valid = 1;
    int tc = imin(t, kV - 1);
    if (tc < kCut0) {
      val = capH[r] - lzH;
    } else if (tc < kCut1) {
      val = (z0[r] - lzH) + (cap0v[r] - lz0);
    } else {
      val = (z1[r] - lzH) + (cap1v[r] - lz1);
    }
  }
  out[r] = val;

  ssum[tid] = val;
  scnt[tid] = valid;
  __syncthreads();
  for (int st = 128; st > 0; st >>= 1) {
    if (tid < st) {
      ssum[tid] += ssum[tid + st];
      scnt[tid] += scnt[tid + st];
    }
    __syncthreads();
  }
  if (tid == 0) {
    blkSum[blockIdx.x] = ssum[0];
    blkCnt[blockIdx.x] = scnt[0];
  }
}

__global__ void loss_kernel(const float* __restrict__ blkSum,
                            const int* __restrict__ blkCnt, int nblk,
                            float* __restrict__ lossOut) {
  if (blockIdx.x == 0 && threadIdx.x == 0) {
    float s = 0.0f;
    int c = 0;
    for (int i = 0; i < nblk; ++i) {
      s += blkSum[i];
      c += blkCnt[i];
    }
    float denom = (float)(c > 0 ? c : 1);
    lossOut[0] = (c > 0) ? (-s / denom) : 0.0f;
  }
}

// ---------------------------------------------------------------------------
extern "C" void kernel_launch(void* const* d_in, const int* in_sizes, int n_in,
                              void* d_out, int out_size, void* d_ws,
                              size_t ws_size, hipStream_t stream) {
  const float* hidden = (const float*)d_in[0];
  const int* target   = (const int*)d_in[1];
  const float* head_w = (const float*)d_in[2];
  const float* proj0  = (const float*)d_in[3];
  const float* tail0  = (const float*)d_in[4];
  const float* proj1  = (const float*)d_in[5];
  const float* tail1  = (const float*)d_in[6];
  float* out = (float*)d_out;  // [0..4095] = output_full, [4096] = loss

  // ---- workspace carve-out (256B aligned) ----
  char* base = (char*)d_ws;
  size_t off = 0;
  auto carve = [&](size_t bytes) -> void* {
    off = (off + 255) & ~(size_t)255;
    void* p = base + off;
    off += bytes;
    return p;
  };
  __bf16* hiddenBF = (__bf16*)carve((size_t)kN * kD * 2);
  __bf16* headwBF  = (__bf16*)carve((size_t)kHeadColsPad * kD * 2);
  __bf16* proj0BF  = (__bf16*)carve((size_t)kK0 * kD * 2);
  __bf16* tail0BF  = (__bf16*)carve((size_t)kT0ColsPad * kK0 * 2);
  __bf16* proj1BF  = (__bf16*)carve((size_t)kK1 * kD * 2);
  __bf16* tail1BF  = (__bf16*)carve((size_t)kT1ColsPad * kK1 * 2);
  __bf16* h0BF     = (__bf16*)carve((size_t)kN * kK0 * 2);
  __bf16* h1BF     = (__bf16*)carve((size_t)kN * kK1 * 2);
  float* pmH = (float*)carve((size_t)kN * kMaxChunk * 4);
  float* psH = (float*)carve((size_t)kN * kMaxChunk * 4);
  float* pm0 = (float*)carve((size_t)kN * kMaxChunk * 4);
  float* ps0 = (float*)carve((size_t)kN * kMaxChunk * 4);
  float* pm1 = (float*)carve((size_t)kN * kMaxChunk * 4);
  float* ps1 = (float*)carve((size_t)kN * kMaxChunk * 4);
  float* capH  = (float*)carve((size_t)kN * 4);
  float* z2000 = (float*)carve((size_t)kN * 4);
  float* z2001 = (float*)carve((size_t)kN * 4);
  float* cap0v = (float*)carve((size_t)kN * 4);
  float* cap1v = (float*)carve((size_t)kN * 4);
  int* capcH = (int*)carve((size_t)kN * 4);
  int* capc0 = (int*)carve((size_t)kN * 4);
  int* capc1 = (int*)carve((size_t)kN * 4);
  float* blkSum = (float*)carve(16 * 4);
  int* blkCnt   = (int*)carve(16 * 4);
  (void)ws_size; (void)n_in; (void)in_sizes; (void)out_size;

  auto cvt = [&](const float* src, __bf16* dst, long sn, long dn) {
    int blocks = (int)((dn + 255) / 256);
    cvt_pad_bf16_kernel<<<blocks, 256, 0, stream>>>(src, dst, sn, dn);
  };
  cvt(hidden, hiddenBF, (long)kN * kD, (long)kN * kD);
  cvt(head_w, headwBF, (long)kHeadCols * kD, (long)kHeadColsPad * kD);
  cvt(proj0, proj0BF, (long)kK0 * kD, (long)kK0 * kD);
  cvt(tail0, tail0BF, (long)kT0Cols * kK0, (long)kT0ColsPad * kK0);
  cvt(proj1, proj1BF, (long)kK1 * kD, (long)kK1 * kD);
  cvt(tail1, tail1BF, (long)kT1Cols * kK1, (long)kT1ColsPad * kK1);

  prep_targets_kernel<<<(kN + 255) / 256, 256, 0, stream>>>(target, capcH,
                                                            capc0, capc1);

  // projections: h0 = hidden @ proj0^T (4096x256), h1 = hidden @ proj1^T
  {
    dim3 grid(kN / 16, (kK0 / 16 + kWaves - 1) / kWaves);
    proj_gemm_kernel<kD><<<grid, kWaves * 32, 0, stream>>>(hiddenBF, proj0BF,
                                                           h0BF, kK0);
  }
  {
    dim3 grid(kN / 16, (kK1 / 16 + kWaves - 1) / kWaves);
    proj_gemm_kernel<kD><<<grid, kWaves * 32, 0, stream>>>(hiddenBF, proj1BF,
                                                           h1BF, kK1);
  }

  // fused GEMM + online softmax partials
  const int ntH = kHeadColsPad / 16;  // 126
  const int ncH = 4;
  const int tpcH = (ntH + ncH - 1) / ncH;
  const int nt0 = kT0ColsPad / 16;    // 500
  const int nc0 = 8;
  const int tpc0 = (nt0 + nc0 - 1) / nc0;
  const int nt1 = kT1ColsPad / 16;    // 2517
  const int nc1 = 20;
  const int tpc1 = (nt1 + nc1 - 1) / nc1;

  {
    dim3 grid(kN / 16, ncH);
    gemm_softmax_kernel<kD, false, true><<<grid, kWaves * 32, 0, stream>>>(
        hiddenBF, headwBF, kHeadCols, ntH, tpcH, capcH, capH, z2000, z2001,
        pmH, psH);
  }
  {
    dim3 grid(kN / 16, nc0);
    gemm_softmax_kernel<kK0, true, false><<<grid, kWaves * 32, 0, stream>>>(
        h0BF, tail0BF, kT0Cols, nt0, tpc0, capc0, cap0v, nullptr, nullptr,
        pm0, ps0);
  }
  {
    dim3 grid(kN / 16, nc1);
    gemm_softmax_kernel<kK1, true, false><<<grid, kWaves * 32, 0, stream>>>(
        h1BF, tail1BF, kT1Cols, nt1, tpc1, capc1, cap1v, nullptr, nullptr,
        pm1, ps1);
  }

  finalize_kernel<<<kN / 256, 256, 0, stream>>>(
      target, capH, z2000, z2001, cap0v, cap1v, pmH, psH, ncH, pm0, ps0, nc0,
      pm1, ps1, nc1, out, blkSum, blkCnt);

  loss_kernel<<<1, 32, 0, stream>>>(blkSum, blkCnt, kN / 256, out + kN);
}
